// MultiheadAttention_81226421502606
// MI455X (gfx1250) — matrix operand order repaired and checked by reading, not compiled, not run on hardware
//
#include <hip/hip_runtime.h>
#include <math.h>

// ---------------------------------------------------------------------------
// MI455X (gfx1250) multihead-attention with RoPE.
// All matmuls via v_wmma_f32_16x16x32_f16. WMMA operands are pre-packed ONCE
// into fragment-major f16 global buffers (L2-resident on the 192MB L2), so
// GEMM/attention inner loops are pure global_load_b128 -> v_wmma. Fragment
// loads are register double-buffered so each step's loads overlap the
// previous step's WMMAs (partial s_wait_loadcnt instead of full drains).
// ---------------------------------------------------------------------------

typedef __attribute__((ext_vector_type(16))) _Float16 v16h;
typedef __attribute__((ext_vector_type(8)))  _Float16 v8h;
typedef __attribute__((ext_vector_type(8)))  float    v8f;

static constexpr int   Bb    = 4;
static constexpr int   Ss    = 1024;
static constexpr int   Ee    = 1024;
static constexpr int   Hh    = 16;
static constexpr int   HDd   = 64;       // E / H
static constexpr float THETA = 10000.0f;

static constexpr int FS = 24;            // LDS fragment row stride (halves)

#define WMMA_F16(A, Bm, C) \
  __builtin_amdgcn_wmma_f32_16x16x32_f16(false, (A), false, (Bm), (short)0, (C), false, false)

__device__ __forceinline__ v16h gload16(const _Float16* p) {   // 32B aligned
  return *(const v16h*)p;
}
__device__ __forceinline__ v16h frag_load_lds(const _Float16* p) {
  v8h lo = *(const v8h*)(p);
  v8h hi = *(const v8h*)(p + 8);
  return __builtin_shufflevector(lo, hi, 0, 1, 2, 3, 4, 5, 6, 7,
                                         8, 9, 10, 11, 12, 13, 14, 15);
}
__device__ __forceinline__ void pack_store(_Float16* p, float a0, float a1) {
  union { _Float16 h[2]; unsigned int u; } pk;
  pk.h[0] = (_Float16)a0;
  pk.h[1] = (_Float16)a1;
  *(unsigned int*)p = pk.u;              // slot even -> 4B aligned
}

// A-fragment (16x32 f16 MxK) inverse map: (m,k) -> lane, slot
__device__ __forceinline__ int aL(int m, int k) { return (m & 15) + 16 * ((k >> 3) & 1); }
__device__ __forceinline__ int aT(int k)        { return (k & 7) + 8 * ((k >> 4) & 1); }
// B-fragment (32x16 f16 KxN) inverse map: (k,n) -> lane, slot
__device__ __forceinline__ int bL(int k, int n) { return (n & 15) + 16 * ((k >> 3) & 1); }

// ---------------------------------------------------------------------------
// Packing kernels: f32 matrices -> fragment-major f16 global buffers.
// Fragment block = 32 lanes x 16 halves = 1024B, 32B contiguous per lane.
// ---------------------------------------------------------------------------
__global__ void pack_a_frag(const float* __restrict__ A, _Float16* __restrict__ Af,
                            int M, int K) {            // A-layout, tiles (M/16, K/32)
  size_t p = (size_t)blockIdx.x * blockDim.x + threadIdx.x;
  size_t total = (size_t)M * K / 2;
  if (p >= total) return;
  int Kp2 = K >> 1;
  int m = (int)(p / Kp2);
  int k = (int)(p % Kp2) * 2;
  const float* src = &A[(size_t)m * K + k];
  int kl = k & 31;
  size_t frag = (size_t)(m >> 4) * (K >> 5) + (k >> 5);
  pack_store(&Af[(frag * 32 + aL(m, kl)) * 16 + aT(kl)], src[0], src[1]);
}

__global__ void pack_b_frag(const float* __restrict__ W, _Float16* __restrict__ Wf,
                            int N, int K) {            // B-layout from W[N,K] (-> W^T)
  size_t p = (size_t)blockIdx.x * blockDim.x + threadIdx.x;
  size_t total = (size_t)N * K / 2;
  if (p >= total) return;
  int Kp2 = K >> 1;
  int n = (int)(p / Kp2);
  int k = (int)(p % Kp2) * 2;
  const float* src = &W[(size_t)n * K + k];
  int kl = k & 31;
  size_t frag = (size_t)(n >> 4) * (K >> 5) + (k >> 5);
  pack_store(&Wf[(frag * 32 + bL(kl, n)) * 16 + aT(kl)], src[0], src[1]);
}

// Q -> A-fragment-major per (hb, s-tile, d-half): Qf[hb][s/16][2][32][16]
__global__ void pack_q_frag(const float* __restrict__ Qp, _Float16* __restrict__ Qf) {
  size_t p = (size_t)blockIdx.x * blockDim.x + threadIdx.x;   // 64*1024*32 pairs
  if (p >= (size_t)(Hh * Bb) * Ss * 32) return;
  int hb = (int)(p >> 15), rem = (int)(p & 32767);
  int s = rem >> 5, d = (rem & 31) * 2;
  int hh = hb / Bb, bbb = hb % Bb;
  const float* src = &Qp[((size_t)bbb * Ss + s) * Ee + hh * HDd + d];
  int dl = d & 31;
  size_t frag = ((size_t)hb * (Ss / 16) + (s >> 4)) * 2 + (d >> 5);
  pack_store(&Qf[(frag * 32 + aL(s, dl)) * 16 + aT(dl)], src[0], src[1]);
}

// K -> B-fragment-major (scores B = K^T): Kf[hb][s/16][2][32][16]
__global__ void pack_k_frag(const float* __restrict__ Kp, _Float16* __restrict__ Kf) {
  size_t p = (size_t)blockIdx.x * blockDim.x + threadIdx.x;   // 64*1024*32 pairs
  if (p >= (size_t)(Hh * Bb) * Ss * 32) return;
  int hb = (int)(p >> 15), rem = (int)(p & 32767);
  int s = rem >> 5, d = (rem & 31) * 2;
  int hh = hb / Bb, bbb = hb % Bb;
  const float* src = &Kp[((size_t)bbb * Ss + s) * Ee + hh * HDd + d];
  int dl = d & 31;
  size_t frag = ((size_t)hb * (Ss / 16) + (s >> 4)) * 2 + (d >> 5);
  pack_store(&Kf[(frag * 32 + bL(dl, s)) * 16 + aT(dl)], src[0], src[1]);
}

// V -> B-fragment-major (AV B = V): Vf[hb][s/32][4][32][16], pairs along s
__global__ void pack_v_frag(const float* __restrict__ Vp, _Float16* __restrict__ Vf) {
  size_t p = (size_t)blockIdx.x * blockDim.x + threadIdx.x;   // 64*512*64 pairs
  if (p >= (size_t)(Hh * Bb) * (Ss / 2) * HDd) return;
  int hb = (int)(p >> 15), rem = (int)(p & 32767);
  int s = (rem >> 6) * 2, d = rem & 63;
  int hh = hb / Bb, bbb = hb % Bb;
  size_t g = ((size_t)bbb * Ss + s) * Ee + hh * HDd + d;
  int kk2 = s & 31;
  size_t frag = ((size_t)hb * (Ss / 32) + (s >> 5)) * 4 + (d >> 4);
  pack_store(&Vf[(frag * 32 + ((d & 15) + 16 * ((kk2 >> 3) & 1))) * 16 + aT(kk2)],
             Vp[g], Vp[g + Ee]);
}

// ---------------------------------------------------------------------------
// LDS-free, barrier-free GEMM (M=4096, N=K=1024), register double-buffered.
// 128 threads (4 waves); each wave: one 16-row x 64-col strip.
// ---------------------------------------------------------------------------
__global__ __launch_bounds__(128)
void gemm_frag_wmma(const _Float16* __restrict__ Af, const _Float16* __restrict__ Wf,
                    const float* __restrict__ bias, float* __restrict__ C) {
  constexpr int K32 = Ee >> 5;           // 32 k-steps
  const int tid  = threadIdx.x;
  const int lane = tid & 31;
  const int wv   = tid >> 5;
  const int m16  = blockIdx.y * 4 + wv;
  const int n16b = blockIdx.x * 4;

  const int nn     = lane & 15;
  const int rowoff = (lane >> 4) * 8;

  const _Float16* ap  = Af + ((size_t)m16 * K32 * 32 + lane) * 16;
  const _Float16* bp0 = Wf + ((size_t)(n16b + 0) * K32 * 32 + lane) * 16;
  const _Float16* bp1 = Wf + ((size_t)(n16b + 1) * K32 * 32 + lane) * 16;
  const _Float16* bp2 = Wf + ((size_t)(n16b + 2) * K32 * 32 + lane) * 16;
  const _Float16* bp3 = Wf + ((size_t)(n16b + 3) * K32 * 32 + lane) * 16;

  v8f acc[4] = {};

  v16h a_c  = gload16(ap);
  v16h b0_c = gload16(bp0);
  v16h b1_c = gload16(bp1);
  v16h b2_c = gload16(bp2);
  v16h b3_c = gload16(bp3);

  for (int k2 = 0; k2 < K32 - 1; ++k2) {
    const size_t off = (size_t)(k2 + 1) * 512;       // next fragment block
    v16h a_n  = gload16(ap  + off);
    v16h b0_n = gload16(bp0 + off);
    v16h b1_n = gload16(bp1 + off);
    v16h b2_n = gload16(bp2 + off);
    v16h b3_n = gload16(bp3 + off);
    acc[0] = WMMA_F16(a_c, b0_c, acc[0]);
    acc[1] = WMMA_F16(a_c, b1_c, acc[1]);
    acc[2] = WMMA_F16(a_c, b2_c, acc[2]);
    acc[3] = WMMA_F16(a_c, b3_c, acc[3]);
    a_c = a_n; b0_c = b0_n; b1_c = b1_n; b2_c = b2_n; b3_c = b3_n;
  }
  acc[0] = WMMA_F16(a_c, b0_c, acc[0]);
  acc[1] = WMMA_F16(a_c, b1_c, acc[1]);
  acc[2] = WMMA_F16(a_c, b2_c, acc[2]);
  acc[3] = WMMA_F16(a_c, b3_c, acc[3]);

#pragma unroll
  for (int nt = 0; nt < 4; ++nt) {
#pragma unroll
    for (int r = 0; r < 8; ++r) {
      int row = m16 * 16 + rowoff + r;
      int col = (n16b + nt) * 16 + nn;
      C[(size_t)row * Ee + col] = acc[nt][r] + bias[col];
    }
  }
}

// ---------------------------------------------------------------------------
// RoPE in place over full embed dim: pairs (2c,2c+1), angle = idx[b,s]*freq(c)
// ---------------------------------------------------------------------------
__global__ void rope_inplace(float* __restrict__ X, const int* __restrict__ idx) {
  size_t p = (size_t)blockIdx.x * blockDim.x + threadIdx.x;
  const size_t total = (size_t)Bb * Ss * (Ee / 2);
  if (p >= total) return;
  int    c  = (int)(p % (Ee / 2));
  size_t bs = p / (Ee / 2);
  float freq = __expf(-(2.0f * (float)c / (float)Ee) * __logf(THETA));
  float ang  = (float)idx[bs] * freq;
  float sn, cs;
  __sincosf(ang, &sn, &cs);
  float x0 = X[2 * p], x1 = X[2 * p + 1];
  X[2 * p]     = x0 * cs - x1 * sn;
  X[2 * p + 1] = x0 * sn + x1 * cs;
}

__global__ void zero_f32(float* __restrict__ p, size_t n) {
  size_t i = (size_t)blockIdx.x * blockDim.x + threadIdx.x;
  if (i < n) p[i] = 0.0f;
}

// ---------------------------------------------------------------------------
// Attention: one wave per (16-query tile, flat head-batch n = h*B + b).
// Operands from packed f16 global (L2); only the 16x1024 score strip in LDS.
// Both WMMA loops register double-buffered; no barriers inside loops.
// ---------------------------------------------------------------------------
__global__ __launch_bounds__(32)
void attn_wmma(const _Float16* __restrict__ Qf, const _Float16* __restrict__ Kf,
               const _Float16* __restrict__ Vf, const unsigned char* __restrict__ kpm,
               float* __restrict__ headOut, float* __restrict__ wOut) {
  __shared__ __align__(16) _Float16 WF[32][32][FS];  // weights, fragment-major

  const int lane  = threadIdx.x;
  const int mtile = blockIdx.x;        // 0..63
  const int nidx  = blockIdx.y;        // flat head-major n = h*B + b
  const int hh    = nidx / Bb;
  const int bb    = nidx % Bb;
  const int bm    = nidx / Hh;         // reference repeat_interleave mask quirk
  const int q0    = mtile * 16;

  const int nn     = lane & 15;
  const int rowoff = (lane >> 4) * 8;

  // Q fragments (d = 0..31 / 32..63): direct packed global loads
  const _Float16* qf = Qf + (((size_t)nidx * (Ss / 16) + mtile) * 2 * 32 + lane) * 16;
  v16h aq0 = gload16(qf);
  v16h aq1 = gload16(qf + 512);

  // ---- scores (double-buffered K fragments) ----
  const _Float16* kfb = Kf + ((size_t)nidx * (Ss / 16) * 2 * 32 + lane) * 16;

#define SCORE_WB(JJ, SC)                                                      \
  _Pragma("unroll")                                                           \
  for (int r = 0; r < 8; ++r) {                                               \
    int   qm   = rowoff + r;                                                  \
    int   kcol = (JJ) * 16 + nn;                                              \
    float vv   = (SC)[r] * 0.125f; /* 1/sqrt(HD) */                           \
    bool  msk  = (kcol > q0 + qm) || (kpm[(size_t)bm * Ss + kcol] != 0);      \
    int   j2i = kcol >> 5, kki = kcol & 31;                                   \
    WF[j2i][aL(qm, kki)][aT(kki)] = msk ? (_Float16)(-INFINITY) : (_Float16)vv; \
  }

  v16h bk0_c = gload16(kfb);
  v16h bk1_c = gload16(kfb + 512);
  for (int j = 0; j < Ss / 16 - 1; ++j) {
    const size_t off = (size_t)(j + 1) * 1024;
    v16h bk0_n = gload16(kfb + off);
    v16h bk1_n = gload16(kfb + off + 512);
    v8f sc = {};
    sc = WMMA_F16(aq0, bk0_c, sc);
    sc = WMMA_F16(aq1, bk1_c, sc);
    SCORE_WB(j, sc)
    bk0_c = bk0_n;
    bk1_c = bk1_n;
  }
  {
    v8f sc = {};
    sc = WMMA_F16(aq0, bk0_c, sc);
    sc = WMMA_F16(aq1, bk1_c, sc);
    SCORE_WB(Ss / 16 - 1, sc)
  }
#undef SCORE_WB
  __syncthreads();

  // ---- softmax: rowwise max & sum; store UNNORMALIZED exp back (f16) ----
  const int row  = lane & 15;
  const int half = lane >> 4;
  const int Lrow = row + 16 * half;              // == lane

  float mx = -INFINITY;
  for (int j2 = 0; j2 < 32; ++j2) {
    v8h lo = *(const v8h*)&WF[j2][Lrow][0];
    v8h hi = *(const v8h*)&WF[j2][Lrow][8];
#pragma unroll
    for (int t = 0; t < 8; ++t) {
      mx = fmaxf(mx, (float)lo[t]);
      mx = fmaxf(mx, (float)hi[t]);
    }
  }
  mx = fmaxf(mx, __shfl_xor(mx, 16, 32));

  float sm = 0.0f;
  for (int j2 = 0; j2 < 32; ++j2) {
    v8h lo = *(const v8h*)&WF[j2][Lrow][0];
    v8h hi = *(const v8h*)&WF[j2][Lrow][8];
    v8h elo, ehi;
#pragma unroll
    for (int t = 0; t < 8; ++t) {
      float e0 = __expf((float)lo[t] - mx);
      float e1 = __expf((float)hi[t] - mx);
      sm += e0 + e1;
      elo[t] = (_Float16)e0;
      ehi[t] = (_Float16)e1;
    }
    *(v8h*)&WF[j2][Lrow][0] = elo;
    *(v8h*)&WF[j2][Lrow][8] = ehi;
  }
  sm += __shfl_xor(sm, 16, 32);
  const float inv = 1.0f / sm;

  // mean attention weights -> wOut (normalized here; LDS stays unnormalized)
  {
    float* wrow = wOut + ((size_t)bb * Ss + q0 + row) * Ss;
    const float scale = inv * (1.0f / Hh);
    for (int j2 = 0; j2 < 32; ++j2) {
      v8h lo = *(const v8h*)&WF[j2][Lrow][0];
      v8h hi = *(const v8h*)&WF[j2][Lrow][8];
#pragma unroll
      for (int t = 0; t < 8; ++t) {
        int kk0 = t + 8 * half;            // slots 0..7  -> kk group 0
        int kk1 = t + 8 * half + 16;       // slots 8..15 -> kk group 1
        atomicAdd(&wrow[j2 * 32 + kk0], (float)lo[t] * scale);
        atomicAdd(&wrow[j2 * 32 + kk1], (float)hi[t] * scale);
      }
    }
  }
  __syncthreads();

  // ---- A.V (double-buffered V fragments; A from LDS weight strip) ----
  const _Float16* vfb = Vf + ((size_t)nidx * (Ss / 32) * 4 * 32 + lane) * 16;
  v8f acc[4] = {};

  v16h aw_c  = frag_load_lds(&WF[0][lane][0]);
  v16h bv0_c = gload16(vfb);
  v16h bv1_c = gload16(vfb + 512);
  v16h bv2_c = gload16(vfb + 1024);
  v16h bv3_c = gload16(vfb + 1536);

  for (int j2 = 0; j2 < Ss / 32 - 1; ++j2) {
    const size_t off = (size_t)(j2 + 1) * 2048;
    v16h aw_n  = frag_load_lds(&WF[j2 + 1][lane][0]);
    v16h bv0_n = gload16(vfb + off);
    v16h bv1_n = gload16(vfb + off + 512);
    v16h bv2_n = gload16(vfb + off + 1024);
    v16h bv3_n = gload16(vfb + off + 1536);
    acc[0] = WMMA_F16(aw_c, bv0_c, acc[0]);
    acc[1] = WMMA_F16(aw_c, bv1_c, acc[1]);
    acc[2] = WMMA_F16(aw_c, bv2_c, acc[2]);
    acc[3] = WMMA_F16(aw_c, bv3_c, acc[3]);
    aw_c = aw_n; bv0_c = bv0_n; bv1_c = bv1_n; bv2_c = bv2_n; bv3_c = bv3_n;
  }
  acc[0] = WMMA_F16(aw_c, bv0_c, acc[0]);
  acc[1] = WMMA_F16(aw_c, bv1_c, acc[1]);
  acc[2] = WMMA_F16(aw_c, bv2_c, acc[2]);
  acc[3] = WMMA_F16(aw_c, bv3_c, acc[3]);

#pragma unroll
  for (int nt = 0; nt < 4; ++nt) {
#pragma unroll
    for (int r = 0; r < 8; ++r) {
      int   qm = rowoff + r;
      float iv = __shfl(inv, qm, 32);      // row's 1/sum
      int   d  = nt * 16 + nn;
      headOut[((size_t)bb * Ss + q0 + qm) * Ee + hh * HDd + d] = acc[nt][r] * iv;
    }
  }
}

// ---------------------------------------------------------------------------
extern "C" void kernel_launch(void* const* d_in, const int* in_sizes, int n_in,
                              void* d_out, int out_size, void* d_ws, size_t ws_size,
                              hipStream_t stream) {
  (void)in_sizes; (void)n_in; (void)out_size; (void)ws_size;

  const float* q    = (const float*)d_in[0];
  const float* k    = (const float*)d_in[1];
  const float* v    = (const float*)d_in[2];
  const int*   qidx = (const int*)d_in[3];
  const int*   kidx = (const int*)d_in[4];
  // d_in[5] attn_mask: causal triu(k=1), computed analytically in-kernel.
  const unsigned char* kpm = (const unsigned char*)d_in[6];
  const float* Wq = (const float*)d_in[7];
  const float* bq = (const float*)d_in[8];
  const float* Wk = (const float*)d_in[9];
  const float* bk = (const float*)d_in[10];
  const float* Wv = (const float*)d_in[11];
  const float* bv = (const float*)d_in[12];
  const float* Wo = (const float*)d_in[13];
  const float* bo = (const float*)d_in[14];

  const size_t BSE = (size_t)Bb * Ss * Ee;    // 4,194,304

  // f32 intermediates
  float* Qp = (float*)d_ws;
  float* Kp = Qp + BSE;
  float* Vp = Kp + BSE;
  float* Hp = Vp + BSE;
  // packed f16 fragment buffers
  _Float16* Qf = (_Float16*)(Hp + BSE);
  _Float16* Kf = Qf + BSE;
  _Float16* Vf = Kf + BSE;
  _Float16* Af = Vf + BSE;                    // GEMM A fragments (reused)
  _Float16* Wf = Af + BSE;                    // GEMM W fragments (reused)

  float* outA = (float*)d_out;                // attn_output  [B,S,E]
  float* outW = outA + BSE;                   // attn_weights [B,S,S]

  const int M = Bb * Ss;                      // 4096
  dim3 gGemm(Ee / 64, M / 64);                // 16 x 64
  const unsigned gPackA = (unsigned)(((size_t)M * Ee / 2 + 255) / 256);
  const unsigned gPackW = (unsigned)(((size_t)Ee * Ee / 2 + 255) / 256);
  const unsigned gPackF = (unsigned)(((size_t)(Hh * Bb) * Ss * 32 + 255) / 256);

  // QKV projections (pack -> fragment GEMM)
  pack_a_frag<<<gPackA, 256, 0, stream>>>(q, Af, M, Ee);
  pack_b_frag<<<gPackW, 256, 0, stream>>>(Wq, Wf, Ee, Ee);
  gemm_frag_wmma<<<gGemm, 128, 0, stream>>>(Af, Wf, bq, Qp);

  pack_a_frag<<<gPackA, 256, 0, stream>>>(k, Af, M, Ee);
  pack_b_frag<<<gPackW, 256, 0, stream>>>(Wk, Wf, Ee, Ee);
  gemm_frag_wmma<<<gGemm, 128, 0, stream>>>(Af, Wf, bk, Kp);

  pack_a_frag<<<gPackA, 256, 0, stream>>>(v, Af, M, Ee);
  pack_b_frag<<<gPackW, 256, 0, stream>>>(Wv, Wf, Ee, Ee);
  gemm_frag_wmma<<<gGemm, 128, 0, stream>>>(Af, Wf, bv, Vp);

  // RoPE (in place, f32)
  const size_t pairs = BSE / 2;
  rope_inplace<<<(unsigned)((pairs + 255) / 256), 256, 0, stream>>>(Qp, qidx);
  rope_inplace<<<(unsigned)((pairs + 255) / 256), 256, 0, stream>>>(Kp, kidx);

  // pack attention operands (fragment-major f16, L2-resident)
  pack_q_frag<<<gPackF, 256, 0, stream>>>(Qp, Qf);
  pack_k_frag<<<gPackF, 256, 0, stream>>>(Kp, Kf);
  pack_v_frag<<<gPackF, 256, 0, stream>>>(Vp, Vf);

  // attn_weights accumulator must start at zero (d_out is poisoned)
  zero_f32<<<(unsigned)((BSE + 255) / 256), 256, 0, stream>>>(outW, BSE);

  attn_wmma<<<dim3(Ss / 16, Hh * Bb), 32, 0, stream>>>(Qf, Kf, Vf, kpm, Hp, outW);

  // output projection
  pack_a_frag<<<gPackA, 256, 0, stream>>>(Hp, Af, M, Ee);
  pack_b_frag<<<gPackW, 256, 0, stream>>>(Wo, Wf, Ee, Ee);
  gemm_frag_wmma<<<gGemm, 128, 0, stream>>>(Af, Wf, bo, outA);
}
